// _MPLBackbone_33767032881755
// MI455X (gfx1250) — matrix operand order repaired and checked by reading, not compiled, not run on hardware
//
#include <hip/hip_runtime.h>
#include <hip/hip_bf16.h>
#include <math.h>

typedef __bf16 bf16_t;
typedef __attribute__((ext_vector_type(16))) __bf16 v16bf;
typedef __attribute__((ext_vector_type(8)))  float  v8f;

#define N_NODES  50000
#define N_EDGES  800000
#define N_GRAPHS 512

// ---------------------------------------------------------------------------
// helpers
// ---------------------------------------------------------------------------
__device__ inline float sigm_(float x) { return 1.0f / (1.0f + __expf(-x)); }

__device__ inline void atomicMaxF(float* addr, float val) {
    unsigned int* ua = (unsigned int*)addr;
    unsigned int old = *ua;
    while (true) {
        float f = __uint_as_float(old);
        if (f >= val) break;
        unsigned int assumed = old;
        old = atomicCAS(ua, assumed, __float_as_uint(val));
        if (old == assumed) break;
    }
}

__global__ void fillf(float* __restrict__ p, float v, size_t n) {
    size_t i = (size_t)blockIdx.x * blockDim.x + threadIdx.x;
    if (i < n) p[i] = v;
}

// ---------------------------------------------------------------------------
// Weight packing: f32 [K, COUT] row-major  ->  bf16 in WMMA B-fragment order:
//   Wp[((nt*KT + kt)*32 + lane)*16 + h] = W[(kt*32 + (lane&16?8:0) + (h<8?h:8+h))*COUT
//                                           + nt*16 + (lane&15)]
// (B 32x16: lanes 0-15 hold K=k0..k0+7,k0+16..k0+23; lanes 16-31 K=k0+8..k0+15,k0+24..k0+31)
// Rows k >= K are zero-padded (used when CIN=16 pads to K=32).
// ---------------------------------------------------------------------------
__global__ void pack_w(const float* __restrict__ W, bf16_t* __restrict__ Wp,
                       int K, int KP, int cout) {
    int idx = blockIdx.x * blockDim.x + threadIdx.x;
    int total = (cout / 16) * (KP / 32) * 512;
    if (idx >= total) return;
    int h    = idx & 15;
    int lane = (idx >> 4) & 31;
    int kt   = (idx >> 9) % (KP / 32);
    int nt   = idx / ((KP / 32) * 512);
    int n = nt * 16 + (lane & 15);
    int k = kt * 32 + ((lane & 16) ? 8 : 0) + (h < 8 ? h : 16 + (h - 8));
    Wp[idx] = (k < K) ? (bf16_t)W[(size_t)k * cout + n] : (bf16_t)0.f;
}

// ---------------------------------------------------------------------------
// Edge message GEMM:  msg_e = relu( [x_i ; x_j - x_i] @ W + b )
// block = 128 threads (4 waves), 64 edges (4 M-tiles of 16).
// A fragments staged in LDS pre-swizzled to WMMA lane order (contiguous 32B
// per lane -> ds_load_b128 x2); B fragments pre-packed bf16 in global
// (contiguous 32B per lane -> global_load_b128 x2), each B feeds 4 WMMAs.
// Zero C via the inline-constant operand; bias+relu in the epilogue.
// ---------------------------------------------------------------------------
template<int TWO_CIN, int COUT>
__global__ __launch_bounds__(128) void edge_msg_wmma(
    const float* __restrict__ x, const int* __restrict__ src,
    const int* __restrict__ dst, const bf16_t* __restrict__ Wp,
    const float* __restrict__ bias, bf16_t* __restrict__ msg)
{
    constexpr int CIN = TWO_CIN / 2;
    constexpr int KT  = TWO_CIN / 32;
    __shared__ int sidx[64], didx[64];
    __shared__ alignas(32) bf16_t Als[4][KT][32][16];

    const int e0  = blockIdx.x * 64;
    const int tid = threadIdx.x;

    // hint: packed weights are re-read by every block; pull toward caches
    __builtin_prefetch(Wp + (size_t)tid * 64, 0, 3);

    if (tid < 64) { sidx[tid] = src[e0 + tid]; didx[tid] = dst[e0 + tid]; }
    __syncthreads();

    // stage gathered features directly in WMMA A-fragment order:
    // A 16x32: lane m holds K = kb..kb+7 (halves 0-7) and kb+16..kb+23
    // (halves 8-15), kb = (lane&16 ? 8 : 0)
    for (int idx = tid; idx < 64 * TWO_CIN; idx += 128) {
        int h    = idx & 15;
        int lane = (idx >> 4) & 31;
        int kt   = (idx >> 9) % KT;
        int et   = idx / (KT * 512);
        int m  = lane & 15;
        int kb = (lane & 16) ? 8 : 0;
        int c  = kt * 32 + kb + (h < 8 ? h : 16 + (h - 8));
        int eb = et * 16 + m;
        int i = didx[eb], j = sidx[eb];
        float v;
        if (c < CIN) v = x[(size_t)i * CIN + c];
        else { int cc = c - CIN; v = x[(size_t)j * CIN + cc] - x[(size_t)i * CIN + cc]; }
        ((bf16_t*)Als)[idx] = (bf16_t)v;
    }
    __syncthreads();

    const int lane = tid & 31;
    const int wave = tid >> 5;
    const int m    = lane & 15;
    const int mb   = (lane & 16) ? 8 : 0;

    for (int nt = wave; nt < COUT / 16; nt += 4) {
        const int n = nt * 16 + m;
        v8f acc[4];
        const v8f z = {0.f,0.f,0.f,0.f,0.f,0.f,0.f,0.f};
#pragma unroll
        for (int et = 0; et < 4; ++et) acc[et] = z;
#pragma unroll
        for (int kt = 0; kt < KT; ++kt) {
            v16bf b = *(const v16bf*)(Wp + (((size_t)nt * KT + kt) * 32 + lane) * 16);
#pragma unroll
            for (int et = 0; et < 4; ++et) {
                v16bf a = *(const v16bf*)(&Als[et][kt][lane][0]);
                acc[et] = __builtin_amdgcn_wmma_f32_16x16x32_bf16(
                              false, a, false, b, (short)0, acc[et], false, false);
            }
        }
        const float bn = bias[n];
#pragma unroll
        for (int et = 0; et < 4; ++et) {
#pragma unroll
            for (int r = 0; r < 8; ++r) {
                float v = fmaxf(acc[et][r] + bn, 0.f);
                msg[(size_t)(e0 + et * 16 + mb + r) * COUT + n] = (bf16_t)v;
            }
        }
    }
}

// ---------------------------------------------------------------------------
// Node GEMM:  x_t = relu( x @ W + b )  — same tiling, 64 rows per block,
// row guards only on stage/store (wave control flow stays uniform for WMMA).
// ---------------------------------------------------------------------------
template<int CIN, int COUT>
__global__ __launch_bounds__(128) void node_gemm_relu(
    const float* __restrict__ x, const bf16_t* __restrict__ Wp,
    const float* __restrict__ bias, float* __restrict__ out)
{
    constexpr int KP = (CIN + 31) & ~31;
    constexpr int KT = KP / 32;
    __shared__ alignas(32) bf16_t Als[4][KT][32][16];

    const int r0  = blockIdx.x * 64;
    const int tid = threadIdx.x;

    __builtin_prefetch(Wp + (size_t)tid * 64, 0, 3);

    for (int idx = tid; idx < 64 * KP; idx += 128) {
        int h    = idx & 15;
        int lane = (idx >> 4) & 31;
        int kt   = (idx >> 9) % KT;
        int et   = idx / (KT * 512);
        int m  = lane & 15;
        int kb = (lane & 16) ? 8 : 0;
        int c  = kt * 32 + kb + (h < 8 ? h : 16 + (h - 8));
        int row = r0 + et * 16 + m;
        float v = (row < N_NODES && c < CIN) ? x[(size_t)row * CIN + c] : 0.f;
        ((bf16_t*)Als)[idx] = (bf16_t)v;
    }
    __syncthreads();

    const int lane = tid & 31;
    const int wave = tid >> 5;
    const int m    = lane & 15;
    const int mb   = (lane & 16) ? 8 : 0;

    for (int nt = wave; nt < COUT / 16; nt += 4) {
        const int n = nt * 16 + m;
        v8f acc[4];
        const v8f z = {0.f,0.f,0.f,0.f,0.f,0.f,0.f,0.f};
#pragma unroll
        for (int et = 0; et < 4; ++et) acc[et] = z;
#pragma unroll
        for (int kt = 0; kt < KT; ++kt) {
            v16bf b = *(const v16bf*)(Wp + (((size_t)nt * KT + kt) * 32 + lane) * 16);
#pragma unroll
            for (int et = 0; et < 4; ++et) {
                v16bf a = *(const v16bf*)(&Als[et][kt][lane][0]);
                acc[et] = __builtin_amdgcn_wmma_f32_16x16x32_bf16(
                              false, a, false, b, (short)0, acc[et], false, false);
            }
        }
        const float bn = bias[n];
#pragma unroll
        for (int et = 0; et < 4; ++et) {
#pragma unroll
            for (int r = 0; r < 8; ++r) {
                int row = r0 + et * 16 + mb + r;
                if (row < N_NODES)
                    out[(size_t)row * COUT + n] = fmaxf(acc[et][r] + bn, 0.f);
            }
        }
    }
}

// ---------------------------------------------------------------------------
// Per-edge attention logit: w = score( tanh(alpha(x_i)) * tanh(beta(msg_e)) )
// ---------------------------------------------------------------------------
__global__ void edge_attn(
    const float* __restrict__ x, const int* __restrict__ dst,
    const float* __restrict__ Wa, const float* __restrict__ ba,
    const float* __restrict__ Wb, const float* __restrict__ bb,
    const float* __restrict__ Ws, const float* __restrict__ bs,
    const bf16_t* __restrict__ msg, float* __restrict__ wl,
    int cin, int cout)
{
    int e = blockIdx.x * blockDim.x + threadIdx.x;
    if (e >= N_EDGES) return;
    int i = dst[e];

    float a[16];
#pragma unroll
    for (int k = 0; k < 16; ++k) a[k] = ba[k];
    const float* xr = x + (size_t)i * cin;
    for (int c = 0; c < cin; ++c) {
        float xi = xr[c];
        const float* wr = Wa + (size_t)c * 16;
#pragma unroll
        for (int k = 0; k < 16; ++k) a[k] += xi * wr[k];
    }
#pragma unroll
    for (int k = 0; k < 16; ++k) a[k] = tanhf(a[k]);

    float bv[16];
#pragma unroll
    for (int k = 0; k < 16; ++k) bv[k] = bb[k];
    const bf16_t* mr = msg + (size_t)e * cout;
    for (int c = 0; c < cout; ++c) {
        float mv = (float)mr[c];
        const float* wr = Wb + (size_t)c * 16;
#pragma unroll
        for (int k = 0; k < 16; ++k) bv[k] += mv * wr[k];
    }
    float s = bs[0];
#pragma unroll
    for (int k = 0; k < 16; ++k) s += a[k] * tanhf(bv[k]) * Ws[k];
    wl[e] = s;
}

// ---------------------------------------------------------------------------
// segment softmax pieces (generic over segment array)
// ---------------------------------------------------------------------------
__global__ void seg_init(float* __restrict__ m, float* __restrict__ s, int n) {
    int i = blockIdx.x * blockDim.x + threadIdx.x;
    if (i < n) { m[i] = -1e30f; s[i] = 0.f; }
}
__global__ void grp_max(const float* __restrict__ v, const int* __restrict__ seg,
                        float* __restrict__ m, int n) {
    int i = blockIdx.x * blockDim.x + threadIdx.x;
    if (i < n) atomicMaxF(&m[seg[i]], v[i]);
}
__global__ void grp_expsum(const float* __restrict__ v, const int* __restrict__ seg,
                           const float* __restrict__ m, float* __restrict__ s,
                           float* __restrict__ coef, int n) {
    int i = blockIdx.x * blockDim.x + threadIdx.x;
    if (i < n) {
        float e = __expf(v[i] - m[seg[i]]);
        coef[i] = e;
        atomicAdd(&s[seg[i]], e);
    }
}
__global__ void grp_norm(float* __restrict__ coef, const int* __restrict__ seg,
                         const float* __restrict__ s, int n) {
    int i = blockIdx.x * blockDim.x + threadIdx.x;
    if (i < n) coef[i] /= (s[seg[i]] + 1e-16f);
}

// msg aggregation at dst: agg[dst[e], c] += msg_e[e, c] * coef[e]
__global__ void scatter_msg(const bf16_t* __restrict__ msg,
                            const float* __restrict__ coef,
                            const int* __restrict__ dst,
                            float* __restrict__ agg, int cout) {
    size_t idx = (size_t)blockIdx.x * blockDim.x + threadIdx.x;
    if (idx >= (size_t)N_EDGES * cout) return;
    int e = (int)(idx / cout), c = (int)(idx % cout);
    atomicAdd(&agg[(size_t)dst[e] * cout + c], (float)msg[idx] * coef[e]);
}

// gated combine + outer relu
__global__ void combine(const float* __restrict__ xt, const float* __restrict__ agg,
                        const float* __restrict__ Wm, const float* __restrict__ bm,
                        const float* __restrict__ Wn, const float* __restrict__ bn,
                        float* __restrict__ out, int cout) {
    int nd = blockIdx.x * blockDim.x + threadIdx.x;
    if (nd >= N_NODES) return;
    const float* xr = xt  + (size_t)nd * cout;
    const float* ar = agg + (size_t)nd * cout;
    float s1 = bm[0], s2 = bn[0];
    for (int c = 0; c < cout; ++c) {
        s1 += xr[c] * Wm[c] + ar[c] * Wm[cout + c];
        s2 += xr[c] * Wn[c] + ar[c] * Wn[cout + c];
    }
    float w1 = sigm_(s1), w2 = sigm_(s2);
    float* orow = out + (size_t)nd * cout;
    for (int c = 0; c < cout; ++c)
        orow[c] = fmaxf(w1 * ar[c] + w2 * xr[c], 0.f);
}

// pooling
__global__ void pool_logit(const float* __restrict__ h, const float* __restrict__ w,
                           const float* __restrict__ b, float* __restrict__ g, int cout) {
    int nd = blockIdx.x * blockDim.x + threadIdx.x;
    if (nd >= N_NODES) return;
    float s = b[0];
    const float* hr = h + (size_t)nd * cout;
    for (int c = 0; c < cout; ++c) s += hr[c] * w[c];
    g[nd] = s;
}
__global__ void pool_scatter(const float* __restrict__ h, const float* __restrict__ coef,
                             const int* __restrict__ batch, float* __restrict__ out,
                             int cout, int coloff) {
    size_t idx = (size_t)blockIdx.x * blockDim.x + threadIdx.x;
    if (idx >= (size_t)N_NODES * cout) return;
    int nd = (int)(idx / cout), c = (int)(idx % cout);
    atomicAdd(&out[(size_t)batch[nd] * 128 + coloff + c], coef[nd] * h[idx]);
}

// ---------------------------------------------------------------------------
// host-side orchestration
// ---------------------------------------------------------------------------
template<int CIN, int COUT>
static void run_layer(const float* xin, float* hout,
                      const int* src, const int* dst,
                      const float* alpha_b, const float* alpha_w,
                      const float* beta_b,  const float* beta_w,
                      const float* msg_b,   const float* msg_w,
                      const float* node_b,  const float* node_w,
                      const float* score_b, const float* score_w,
                      const float* wmsg_b,  const float* wmsg_w,
                      const float* wnode_b, const float* wnode_w,
                      bf16_t* msg, bf16_t* wpm, bf16_t* wpn,
                      float* xt, float* agg,
                      float* wl, float* coef, float* segm, float* segs,
                      hipStream_t stream)
{
    constexpr int TWO = 2 * CIN;
    constexpr int KP  = (CIN + 31) & ~31;

    // pack weights into bf16 WMMA B-fragment order (tiny, L2-resident after)
    pack_w<<<((COUT / 16) * (TWO / 32) * 512 + 255) / 256, 256, 0, stream>>>(
        msg_w, wpm, TWO, TWO, COUT);
    pack_w<<<((COUT / 16) * (KP / 32) * 512 + 255) / 256, 256, 0, stream>>>(
        node_w, wpn, CIN, KP, COUT);

    fillf<<<(unsigned)(((size_t)N_NODES * COUT + 255) / 256), 256, 0, stream>>>(
        agg, 0.f, (size_t)N_NODES * COUT);
    edge_msg_wmma<TWO, COUT><<<N_EDGES / 64, 128, 0, stream>>>(
        xin, src, dst, wpm, msg_b, msg);
    node_gemm_relu<CIN, COUT><<<(N_NODES + 63) / 64, 128, 0, stream>>>(
        xin, wpn, node_b, xt);
    edge_attn<<<(N_EDGES + 127) / 128, 128, 0, stream>>>(
        xin, dst, alpha_w, alpha_b, beta_w, beta_b, score_w, score_b,
        msg, wl, CIN, COUT);
    seg_init<<<(N_NODES + 255) / 256, 256, 0, stream>>>(segm, segs, N_NODES);
    grp_max<<<(N_EDGES + 255) / 256, 256, 0, stream>>>(wl, src, segm, N_EDGES);
    grp_expsum<<<(N_EDGES + 255) / 256, 256, 0, stream>>>(wl, src, segm, segs, coef, N_EDGES);
    grp_norm<<<(N_EDGES + 255) / 256, 256, 0, stream>>>(coef, src, segs, N_EDGES);
    scatter_msg<<<(unsigned)(((size_t)N_EDGES * COUT + 255) / 256), 256, 0, stream>>>(
        msg, coef, dst, agg, COUT);
    combine<<<(N_NODES + 127) / 128, 128, 0, stream>>>(
        xt, agg, wmsg_w, wmsg_b, wnode_w, wnode_b, hout, COUT);
}

static void run_pool(const float* h, const float* w, const float* b,
                     const int* batch, float* out, int coloff,
                     float* pg, float* pcoef, float* pm, float* psum,
                     hipStream_t stream)
{
    pool_logit<<<(N_NODES + 127) / 128, 128, 0, stream>>>(h, w, b, pg, 64);
    seg_init<<<(N_GRAPHS + 255) / 256, 256, 0, stream>>>(pm, psum, N_GRAPHS);
    grp_max<<<(N_NODES + 255) / 256, 256, 0, stream>>>(pg, batch, pm, N_NODES);
    grp_expsum<<<(N_NODES + 255) / 256, 256, 0, stream>>>(pg, batch, pm, psum, pcoef, N_NODES);
    grp_norm<<<(N_NODES + 255) / 256, 256, 0, stream>>>(pcoef, batch, psum, N_NODES);
    pool_scatter<<<(unsigned)(((size_t)N_NODES * 64 + 255) / 256), 256, 0, stream>>>(
        h, pcoef, batch, out, 64, coloff);
}

extern "C" void kernel_launch(void* const* d_in, const int* in_sizes, int n_in,
                              void* d_out, int out_size, void* d_ws, size_t ws_size,
                              hipStream_t stream)
{
    (void)in_sizes; (void)n_in; (void)out_size; (void)ws_size;

    const float* x     = (const float*)d_in[0];
    const int*   src   = (const int*)d_in[1];           // edge_index[0]
    const int*   dst   = src + N_EDGES;                 // edge_index[1]
    const int*   batch = (const int*)d_in[2];

    // params pytree (dicts flattened with sorted keys):
    // per conv: alpha.b, alpha.w, beta.b, beta.w, msg.b, msg.w, node.b, node.w,
    //           score.b, score.w, w_msg.b, w_msg.w, w_node.b, w_node.w
    const float* cp[4][14];
    for (int cv = 0; cv < 4; ++cv)
        for (int k = 0; k < 14; ++k)
            cp[cv][k] = (const float*)d_in[3 + cv * 14 + k];
    const float* pool1_b = (const float*)d_in[59];
    const float* pool1_w = (const float*)d_in[60];
    const float* pool2_b = (const float*)d_in[61];
    const float* pool2_w = (const float*)d_in[62];

    // workspace carve-up (packed weights first: keeps 32B alignment for b128)
    bf16_t* wsb = (bf16_t*)d_ws;
    bf16_t* wpm = wsb; wsb += 32768;                    // packed msg weights
    bf16_t* wpn = wsb; wsb += 32768;                    // packed node weights
    float* ws   = (float*)wsb;
    float* hA   = ws; ws += (size_t)N_NODES * 128;
    float* hB   = ws; ws += (size_t)N_NODES * 128;
    float* xt   = ws; ws += (size_t)N_NODES * 128;
    float* agg  = ws; ws += (size_t)N_NODES * 128;
    float* wl   = ws; ws += N_EDGES;
    float* coef = ws; ws += N_EDGES;
    float* segm = ws; ws += N_NODES;
    float* segs = ws; ws += N_NODES;
    float* pg   = ws; ws += N_NODES;
    float* pcf  = ws; ws += N_NODES;
    float* pm   = ws; ws += N_GRAPHS;
    float* psum = ws; ws += N_GRAPHS;
    bf16_t* msg = (bf16_t*)ws;                          // E * 128 bf16

    float* out = (float*)d_out;
    fillf<<<(unsigned)(((size_t)N_GRAPHS * 128 + 255) / 256), 256, 0, stream>>>(
        out, 0.f, (size_t)N_GRAPHS * 128);

#define CONV_ARGS(i) \
    cp[i][0], cp[i][1], cp[i][2], cp[i][3], cp[i][4], cp[i][5], cp[i][6], \
    cp[i][7], cp[i][8], cp[i][9], cp[i][10], cp[i][11], cp[i][12], cp[i][13]

    run_layer<16, 128>(x,  hA, src, dst, CONV_ARGS(0),
                       msg, wpm, wpn, xt, agg, wl, coef, segm, segs, stream);
    run_layer<128, 64>(hA, hB, src, dst, CONV_ARGS(1),
                       msg, wpm, wpn, xt, agg, wl, coef, segm, segs, stream);
    run_pool(hB, pool1_w, pool1_b, batch, out, 0,  pg, pcf, pm, psum, stream);
    run_layer<64, 64>(hB, hA, src, dst, CONV_ARGS(2),
                      msg, wpm, wpn, xt, agg, wl, coef, segm, segs, stream);
    run_layer<64, 64>(hA, hB, src, dst, CONV_ARGS(3),
                      msg, wpm, wpn, xt, agg, wl, coef, segm, segs, stream);
    run_pool(hB, pool2_w, pool2_b, batch, out, 64, pg, pcf, pm, psum, stream);
#undef CONV_ARGS
}
